// VoiceCloningModel_24077586662119
// MI455X (gfx1250) — compile-verified, matrix operand order checked by reading, not run on hardware
//
#include <hip/hip_runtime.h>
#include <hip/hip_bf16.h>

// ============================================================================
// VoiceCloningModel forward for MI455X (gfx1250, wave32, WMMA).
//
// All GEMMs use v_wmma_f32_16x16x32_bf16 (bf16 in, f32 accumulate) in "NT"
// form: C[m,n] = sum_k A[m,k]*B[n,k]  (matches x@W.T and Q*K^T / P*V^T).
//
// wide kernel  (N%128==0): 128x128 macro-tile, wave owns 32x64 -> 8 WMMAs
//              per 32-wide K chunk (2 A-frags x 4 B-frags), double-buffered
//              LDS, register prefetch of next chunk, 2 barriers per chunk.
// fast kernel  (N%64==0):  128x64 macro-tile, wave owns 32x32 -> 4 WMMAs.
// guarded      (odd/tiny): 64x32 tile, bounds-checked (M=8 MLPs, N=80 mel,
//              K=80 voc1, logits).
// ============================================================================

typedef unsigned short u16t;
typedef __attribute__((ext_vector_type(16))) __bf16 v16bf;
typedef __attribute__((ext_vector_type(8)))  float  v8f;

struct alignas(16) U4 { unsigned a, b, c, d; };
union FragU { v16bf v; U4 q[2]; };

static __device__ __forceinline__ u16t f2bf(float f) {
  unsigned u = __float_as_uint(f);
  unsigned r = u + 0x7FFFu + ((u >> 16) & 1u);   // round-to-nearest-even
  return (u16t)(r >> 16);
}

#define WMMA_BF16(a, b, c) \
  __builtin_amdgcn_wmma_f32_16x16x32_bf16(false, (a), false, (b), (short)0, (c), false, false)

// ----------------------------------------------------------------------------
// WIDE batched NT GEMM: M%128==0, N%128==0, K%32==0.  z = zb*H + zh.
// ----------------------------------------------------------------------------
__global__ __launch_bounds__(256) void gemm_wide_wmma(
    const u16t* __restrict__ A, int lda, long sAb, long sAh,
    const u16t* __restrict__ Bm, int ldb, long sBb, long sBh,
    float* __restrict__ C, u16t* __restrict__ Cbf, int ldc, long sCb, long sCh,
    int M, int N, int K, const float* __restrict__ bias, int relu, int H)
{
  __shared__ u16t As[2][128][40];   // +8 pad: rows stay 16B aligned
  __shared__ u16t Bs[2][128][40];

  int z = blockIdx.z;
  int zb = z / H, zh = z - zb * H;
  A  += zb * sAb + zh * sAh;
  Bm += zb * sBb + zh * sBh;
  long coff = zb * sCb + zh * sCh;

  int m0 = blockIdx.y * 128;
  int n0 = blockIdx.x * 128;

  int tid  = threadIdx.x;
  int wave = tid >> 5;        // 0..7
  int lane = tid & 31;
  int wm = wave >> 1;         // 0..3 : M sub-tile (32 rows)
  int wn = wave & 1;          // 0..1 : N sub-tile (64 cols)
  int hl = lane >> 4;
  int l  = lane & 15;

  // staging: A tile 128x32 and B tile 128x32, 2 x 16B each per thread
  int sr  = tid >> 2;           // 0..63
  int scg = (tid & 3) * 8;      // 0,8,16,24

  const u16t* pA0 = A  + (long)(m0 + sr) * lda + scg;
  const u16t* pA1 = pA0 + 64L * lda;
  const u16t* pB0 = Bm + (long)(n0 + sr) * ldb + scg;
  const u16t* pB1 = pB0 + 64L * ldb;

  U4 ra0 = *(const U4*)pA0;
  U4 ra1 = *(const U4*)pA1;
  U4 rb0 = *(const U4*)pB0;
  U4 rb1 = *(const U4*)pB1;

  v8f acc[2][4] = {};

  int cur = 0;
  for (int kc = 0; ; ) {
    *(U4*)&As[cur][sr][scg]      = ra0;
    *(U4*)&As[cur][sr + 64][scg] = ra1;
    *(U4*)&Bs[cur][sr][scg]      = rb0;
    *(U4*)&Bs[cur][sr + 64][scg] = rb1;
    __syncthreads();

    kc += 32;
    bool more = kc < K;
    if (more) {   // prefetch next chunk into registers while WMMAs run
      pA0 += 32; pA1 += 32; pB0 += 32; pB1 += 32;
      ra0 = *(const U4*)pA0;
      ra1 = *(const U4*)pA1;
      rb0 = *(const U4*)pB0;
      rb1 = *(const U4*)pB1;
      __builtin_prefetch(pA0 + 32, 0, 1);
      __builtin_prefetch(pB0 + 32, 0, 1);
    }

    // fragments (ISA 16-bit layout: lanes 0-15 K{0..7,16..23},
    // lanes 16-31 K{8..15,24..31}; B mirrors A with N for M)
    FragU a0, a1, b0, b1, b2, b3;
    const u16t* ap0 = &As[cur][wm * 32 + l][hl * 8];
    const u16t* ap1 = ap0 + 16 * 40;
    const u16t* bp0 = &Bs[cur][wn * 64 + l][hl * 8];
    a0.q[0] = *(const U4*)ap0;              a0.q[1] = *(const U4*)(ap0 + 16);
    a1.q[0] = *(const U4*)ap1;              a1.q[1] = *(const U4*)(ap1 + 16);
    b0.q[0] = *(const U4*)bp0;              b0.q[1] = *(const U4*)(bp0 + 16);
    b1.q[0] = *(const U4*)(bp0 + 16 * 40);  b1.q[1] = *(const U4*)(bp0 + 16 * 40 + 16);
    b2.q[0] = *(const U4*)(bp0 + 32 * 40);  b2.q[1] = *(const U4*)(bp0 + 32 * 40 + 16);
    b3.q[0] = *(const U4*)(bp0 + 48 * 40);  b3.q[1] = *(const U4*)(bp0 + 48 * 40 + 16);

    acc[0][0] = WMMA_BF16(a0.v, b0.v, acc[0][0]);
    acc[0][1] = WMMA_BF16(a0.v, b1.v, acc[0][1]);
    acc[0][2] = WMMA_BF16(a0.v, b2.v, acc[0][2]);
    acc[0][3] = WMMA_BF16(a0.v, b3.v, acc[0][3]);
    acc[1][0] = WMMA_BF16(a1.v, b0.v, acc[1][0]);
    acc[1][1] = WMMA_BF16(a1.v, b1.v, acc[1][1]);
    acc[1][2] = WMMA_BF16(a1.v, b2.v, acc[1][2]);
    acc[1][3] = WMMA_BF16(a1.v, b3.v, acc[1][3]);

    if (!more) break;
    __syncthreads();
    cur ^= 1;
  }

  #pragma unroll
  for (int im = 0; im < 2; ++im) {
    #pragma unroll
    for (int in = 0; in < 4; ++in) {
      int n = n0 + wn * 64 + in * 16 + l;
      float bv = bias ? bias[n] : 0.f;
      #pragma unroll
      for (int j = 0; j < 8; ++j) {
        int m = m0 + wm * 32 + im * 16 + j + hl * 8;
        float v = acc[im][in][j] + bv;
        if (relu) v = v > 0.f ? v : 0.f;
        long o = coff + (long)m * ldc + n;
        if (C)   C[o]   = v;
        if (Cbf) Cbf[o] = f2bf(v);
      }
    }
  }
}

// ----------------------------------------------------------------------------
// FAST batched NT GEMM: M%128==0, N%64==0, K%32==0 (encoder P*V, N=64).
// ----------------------------------------------------------------------------
__global__ __launch_bounds__(256) void gemm_fast_wmma(
    const u16t* __restrict__ A, int lda, long sAb, long sAh,
    const u16t* __restrict__ Bm, int ldb, long sBb, long sBh,
    float* __restrict__ C, u16t* __restrict__ Cbf, int ldc, long sCb, long sCh,
    int M, int N, int K, const float* __restrict__ bias, int relu, int H)
{
  __shared__ u16t As[2][128][40];
  __shared__ u16t Bs[2][64][40];

  int z = blockIdx.z;
  int zb = z / H, zh = z - zb * H;
  A  += zb * sAb + zh * sAh;
  Bm += zb * sBb + zh * sBh;
  long coff = zb * sCb + zh * sCh;

  int m0 = blockIdx.y * 128;
  int n0 = blockIdx.x * 64;

  int tid  = threadIdx.x;
  int wave = tid >> 5;
  int lane = tid & 31;
  int wm = wave >> 1;
  int wn = wave & 1;
  int hl = lane >> 4;
  int l  = lane & 15;

  int sr  = tid >> 2;
  int scg = (tid & 3) * 8;

  const u16t* pA0 = A  + (long)(m0 + sr) * lda + scg;
  const u16t* pA1 = pA0 + 64L * lda;
  const u16t* pB  = Bm + (long)(n0 + sr) * ldb + scg;

  U4 ra0 = *(const U4*)pA0;
  U4 ra1 = *(const U4*)pA1;
  U4 rb  = *(const U4*)pB;

  v8f c00 = {}, c01 = {}, c10 = {}, c11 = {};

  int cur = 0;
  for (int kc = 0; ; ) {
    *(U4*)&As[cur][sr][scg]      = ra0;
    *(U4*)&As[cur][sr + 64][scg] = ra1;
    *(U4*)&Bs[cur][sr][scg]      = rb;
    __syncthreads();

    kc += 32;
    bool more = kc < K;
    if (more) {
      pA0 += 32; pA1 += 32; pB += 32;
      ra0 = *(const U4*)pA0;
      ra1 = *(const U4*)pA1;
      rb  = *(const U4*)pB;
      __builtin_prefetch(pA0 + 32, 0, 1);
      __builtin_prefetch(pB  + 32, 0, 1);
    }

    FragU a0, a1, b0, b1;
    const u16t* ap0 = &As[cur][wm * 32 + l][hl * 8];
    const u16t* ap1 = ap0 + 16 * 40;
    const u16t* bp0 = &Bs[cur][wn * 32 + l][hl * 8];
    const u16t* bp1 = bp0 + 16 * 40;
    a0.q[0] = *(const U4*)ap0;  a0.q[1] = *(const U4*)(ap0 + 16);
    a1.q[0] = *(const U4*)ap1;  a1.q[1] = *(const U4*)(ap1 + 16);
    b0.q[0] = *(const U4*)bp0;  b0.q[1] = *(const U4*)(bp0 + 16);
    b1.q[0] = *(const U4*)bp1;  b1.q[1] = *(const U4*)(bp1 + 16);

    c00 = WMMA_BF16(a0.v, b0.v, c00);
    c01 = WMMA_BF16(a0.v, b1.v, c01);
    c10 = WMMA_BF16(a1.v, b0.v, c10);
    c11 = WMMA_BF16(a1.v, b1.v, c11);

    if (!more) break;
    __syncthreads();
    cur ^= 1;
  }

  v8f accs[2][2] = {{c00, c01}, {c10, c11}};
  #pragma unroll
  for (int im = 0; im < 2; ++im) {
    #pragma unroll
    for (int in = 0; in < 2; ++in) {
      int n = n0 + wn * 32 + in * 16 + l;
      float bv = bias ? bias[n] : 0.f;
      #pragma unroll
      for (int j = 0; j < 8; ++j) {
        int m = m0 + wm * 32 + im * 16 + j + hl * 8;
        float v = accs[im][in][j] + bv;
        if (relu) v = v > 0.f ? v : 0.f;
        long o = coff + (long)m * ldc + n;
        if (C)   C[o]   = v;
        if (Cbf) Cbf[o] = f2bf(v);
      }
    }
  }
}

// ----------------------------------------------------------------------------
// GUARDED NT GEMM (small/odd shapes: M=8 MLPs, N=80 mel, K=80 voc1, logits).
// ----------------------------------------------------------------------------
__global__ __launch_bounds__(256) void gemm_bf16_wmma(
    const u16t* __restrict__ A, int lda, long sAb, long sAh,
    const u16t* __restrict__ Bm, int ldb, long sBb, long sBh,
    float* __restrict__ C, u16t* __restrict__ Cbf, int ldc, long sCb, long sCh,
    int M, int N, int K, const float* __restrict__ bias, int relu, int H)
{
  __shared__ u16t As[64][40];
  __shared__ u16t Bs[32][40];

  int z = blockIdx.z;
  int zb = z / H, zh = z - zb * H;
  A  += zb * sAb + zh * sAh;
  Bm += zb * sBb + zh * sBh;
  long coff = zb * sCb + zh * sCh;

  int m0 = blockIdx.y * 64;
  int n0 = blockIdx.x * 32;

  int tid  = threadIdx.x;
  int wave = tid >> 5;
  int lane = tid & 31;
  int wm = wave >> 1;
  int wn = wave & 1;
  int hl = lane >> 4;
  int l  = lane & 15;

  int ar  = tid >> 2;
  int acg = (tid & 3) * 8;

  v8f acc = {};

  for (int kc = 0; kc < K; kc += 32) {
    __syncthreads();
    {
      int gm = m0 + ar, gk = kc + acg;
      u16t* dst = &As[ar][acg];
      if (gm < M && gk + 8 <= K) {
        *(U4*)dst = *(const U4*)(A + (long)gm * lda + gk);
      } else {
        for (int j = 0; j < 8; ++j)
          dst[j] = (gm < M && gk + j < K) ? A[(long)gm * lda + gk + j] : (u16t)0;
      }
    }
    if (tid < 128) {
      int br = tid >> 2, bcg = (tid & 3) * 8;
      int gn = n0 + br, gk = kc + bcg;
      u16t* dst = &Bs[br][bcg];
      if (gn < N && gk + 8 <= K) {
        *(U4*)dst = *(const U4*)(Bm + (long)gn * ldb + gk);
      } else {
        for (int j = 0; j < 8; ++j)
          dst[j] = (gn < N && gk + j < K) ? Bm[(long)gn * ldb + gk + j] : (u16t)0;
      }
    }
    __syncthreads();

    FragU af, bf;
    const u16t* ap = &As[wm * 16 + l][hl * 8];
    af.q[0] = *(const U4*)ap;
    af.q[1] = *(const U4*)(ap + 16);
    const u16t* bp = &Bs[wn * 16 + l][hl * 8];
    bf.q[0] = *(const U4*)bp;
    bf.q[1] = *(const U4*)(bp + 16);

    acc = WMMA_BF16(af.v, bf.v, acc);
  }

  #pragma unroll
  for (int j = 0; j < 8; ++j) {
    int m = m0 + wm * 16 + j + hl * 8;
    int n = n0 + wn * 16 + l;
    if (m < M && n < N) {
      float v = acc[j];
      if (bias) v += bias[n];
      if (relu) v = v > 0.f ? v : 0.f;
      long o = coff + (long)m * ldc + n;
      if (C)   C[o]   = v;
      if (Cbf) Cbf[o] = f2bf(v);
    }
  }
}

// ---------------------------------------------------------------------------
__global__ void cvt_kernel(const float* __restrict__ s, u16t* __restrict__ db,
                           float* __restrict__ df, long n)
{
  long i = (long)blockIdx.x * 256 + threadIdx.x;
  long stride = (long)gridDim.x * 256;
  for (; i < n; i += stride) {
    float v = s[i];
    db[i] = f2bf(v);
    if (df) df[i] = v;
  }
}

__global__ void embed_kernel(const int* __restrict__ text,
                             const float* __restrict__ emb,
                             float* __restrict__ xf, u16t* __restrict__ xb, int n)
{
  int i = blockIdx.x * 256 + threadIdx.x;
  if (i >= n) return;
  int c = i & 511;          // DT = 512
  int bs = i >> 9;
  float v = emb[text[bs] * 512 + c];
  xf[i] = v;
  xb[i] = f2bf(v);
}

// dst[b,h,d,k] = src[(b*Sk+k)*ld + colOff + h*hd + d]   (V -> V^T per head)
__global__ void vtrans_kernel(const u16t* __restrict__ src, int ld, int colOff,
                              int hd, int Sk, int H, u16t* __restrict__ dst, long n)
{
  long i = (long)blockIdx.x * 256 + threadIdx.x;
  if (i >= n) return;
  int k = (int)(i % Sk);
  long r = i / Sk;
  int d = (int)(r % hd); r /= hd;
  int h = (int)(r % H);
  long b = r / H;
  dst[i] = src[(b * Sk + k) * (long)ld + colOff + h * hd + d];
}

__global__ __launch_bounds__(256) void softmax_kernel(
    const float* __restrict__ S, u16t* __restrict__ P, int len, float scale)
{
  __shared__ float red[256];
  long row = blockIdx.x;
  const float* sr = S + row * (long)len;
  float m = -3.0e38f;
  for (int i = threadIdx.x; i < len; i += 256) m = fmaxf(m, sr[i]);
  red[threadIdx.x] = m; __syncthreads();
  for (int o = 128; o > 0; o >>= 1) {
    if (threadIdx.x < o) red[threadIdx.x] = fmaxf(red[threadIdx.x], red[threadIdx.x + o]);
    __syncthreads();
  }
  m = red[0]; __syncthreads();
  float s = 0.f;
  for (int i = threadIdx.x; i < len; i += 256) s += __expf((sr[i] - m) * scale);
  red[threadIdx.x] = s; __syncthreads();
  for (int o = 128; o > 0; o >>= 1) {
    if (threadIdx.x < o) red[threadIdx.x] += red[threadIdx.x + o];
    __syncthreads();
  }
  float inv = 1.f / red[0];
  for (int i = threadIdx.x; i < len; i += 256)
    P[row * (long)len + i] = f2bf(__expf((sr[i] - m) * scale) * inv);
}

// out = LayerNorm(x + t) * g + b ; writes f32 (may alias x) + bf16
__global__ __launch_bounds__(256) void add_ln_kernel(
    const float* __restrict__ x, const float* __restrict__ t,
    const float* __restrict__ g, const float* __restrict__ b,
    float* __restrict__ of, u16t* __restrict__ ob, int D)
{
  __shared__ float red[256];
  long row = blockIdx.x;
  const float* xr = x + row * (long)D;
  const float* tr = t + row * (long)D;
  float s = 0.f;
  for (int i = threadIdx.x; i < D; i += 256) s += xr[i] + tr[i];
  red[threadIdx.x] = s; __syncthreads();
  for (int o = 128; o > 0; o >>= 1) {
    if (threadIdx.x < o) red[threadIdx.x] += red[threadIdx.x + o];
    __syncthreads();
  }
  float mean = red[0] / D; __syncthreads();
  float vs = 0.f;
  for (int i = threadIdx.x; i < D; i += 256) {
    float d = xr[i] + tr[i] - mean; vs += d * d;
  }
  red[threadIdx.x] = vs; __syncthreads();
  for (int o = 128; o > 0; o >>= 1) {
    if (threadIdx.x < o) red[threadIdx.x] += red[threadIdx.x + o];
    __syncthreads();
  }
  float rstd = rsqrtf(red[0] / D + 1e-5f);
  for (int i = threadIdx.x; i < D; i += 256) {
    float v = (xr[i] + tr[i] - mean) * rstd * g[i] + b[i];
    of[row * (long)D + i] = v;
    ob[row * (long)D + i] = f2bf(v);
  }
}

// comb = concat(tf[B,256,512], spk[B,256] bcast, emo[B,256] bcast) -> bf16
__global__ void comb_kernel(const float* __restrict__ tf,
                            const float* __restrict__ spk,
                            const float* __restrict__ emo,
                            u16t* __restrict__ comb, long n)
{
  long i = (long)blockIdx.x * 256 + threadIdx.x;
  if (i >= n) return;
  int c = (int)(i & 1023);
  long bs = i >> 10;        // b*256 + s
  long b = bs >> 8;
  float v;
  if (c < 512)      v = tf[bs * 512 + c];
  else if (c < 768) v = spk[b * 256 + (c - 512)];
  else              v = emo[b * 256 + (c - 768)];
  comb[i] = f2bf(v);
}

// ============================================================================
// Host orchestration
// ============================================================================
static inline int cdiv(int a, int b) { return (a + b - 1) / b; }

static void gemm(hipStream_t st,
                 const u16t* A, int lda, long sAb, long sAh,
                 const u16t* B, int ldb, long sBb, long sBh,
                 float* C, u16t* Cb, int ldc, long sCb, long sCh,
                 int M, int N, int K, const float* bias, int relu, int Z, int H)
{
  if ((M % 128) == 0 && (N % 128) == 0 && (K % 32) == 0) {
    dim3 g(N / 128, M / 128, Z);
    gemm_wide_wmma<<<g, 256, 0, st>>>(A, lda, sAb, sAh, B, ldb, sBb, sBh,
                                      C, Cb, ldc, sCb, sCh, M, N, K, bias, relu, H);
  } else if ((M % 128) == 0 && (N % 64) == 0 && (K % 32) == 0) {
    dim3 g(N / 64, M / 128, Z);
    gemm_fast_wmma<<<g, 256, 0, st>>>(A, lda, sAb, sAh, B, ldb, sBb, sBh,
                                      C, Cb, ldc, sCb, sCh, M, N, K, bias, relu, H);
  } else {
    dim3 g(cdiv(N, 32), cdiv(M, 64), Z);
    gemm_bf16_wmma<<<g, 256, 0, st>>>(A, lda, sAb, sAh, B, ldb, sBb, sBh,
                                      C, Cb, ldc, sCb, sCh, M, N, K, bias, relu, H);
  }
}

static void cvt(hipStream_t st, const float* s, u16t* db, float* df, long n)
{
  long blocks = (n + 255) / 256; if (blocks > 65535) blocks = 65535;
  cvt_kernel<<<(int)blocks, 256, 0, st>>>(s, db, df, n);
}

extern "C" void kernel_launch(void* const* d_in, const int* in_sizes, int n_in,
                              void* d_out, int out_size, void* d_ws, size_t ws_size,
                              hipStream_t stream)
{
  if (n_in < 207) return;
  // ---- input leaf order (setup_inputs() dict-insertion order, recursive) ----
  // 0 text, 1 speaker_embedding, 2 emotion_embedding, 3 target_mel, then
  // params: emb(4); enc L: 5+L*12 {qkv_w,qkv_b,o_w,o_b,f1w,f1b,f2w,f2b,
  // ln1g,ln1b,ln2g,ln2b}; spk 77..82 {w1,b1,w2,b2,w3,b3}; emo 83..88;
  // emo_cls 89,90; dec L: 91+L*18 {sa_qkv_w,sa_qkv_b,sa_o_w,sa_o_b,ca_qkv_w,
  // ca_qkv_b,ca_o_w,ca_o_b,f1w,f1b,f2w,f2b,ln1g,ln1b,ln2g,ln2b,ln3g,ln3b};
  // dec_out 199,200; voc 201..206.
  auto F = [&](int i) { return (const float*)d_in[i]; };
  const int* text = (const int*)d_in[0];

  // ---- workspace bump allocator ----
  char* p = (char*)d_ws;
  size_t used = 0;
  auto alloc = [&](size_t bytes) {
    void* r = p + used;
    used += (bytes + 255) & ~(size_t)255;
    return r;
  };
  u16t*  wconv  = (u16t*) alloc(4194304UL * 2);   // weight bf16 scratch
  float* xe_f   = (float*)alloc(1048576UL * 4);   // encoder stream f32
  u16t*  xe_b   = (u16t*) alloc(1048576UL * 2);
  u16t*  qkvb   = (u16t*) alloc(12582912UL * 2);  // qkv / q+kv bf16
  u16t*  vt     = (u16t*) alloc(4194304UL * 2);   // V^T per head
  float* scores = (float*)alloc(16777216UL * 4);  // scores f32 (reused as FFN h)
  u16t*  pbuf   = (u16t*) alloc(16777216UL * 2);  // softmax probs bf16
  u16t*  attnb  = (u16t*) alloc(4194304UL * 2);   // attention out bf16
  float* tmpf   = (float*)alloc(4194304UL * 4);   // pre-LN GEMM out f32
  float* xd_f   = (float*)alloc(4194304UL * 4);   // decoder stream f32
  u16t*  xd_b   = (u16t*) alloc(4194304UL * 2);
  u16t*  comb   = (u16t*) alloc(2097152UL * 2);   // encoder memory bf16
  u16t*  sb_in  = (u16t*) alloc(4096UL * 2);
  u16t*  sb1    = (u16t*) alloc(4096UL * 2);
  u16t*  sb2    = (u16t*) alloc(4096UL * 2);
  float* spk_f  = (float*)alloc(2048UL * 4);
  u16t*  eb_in  = (u16t*) alloc(2048UL * 2);
  u16t*  eb1    = (u16t*) alloc(2048UL * 2);
  u16t*  eb2    = (u16t*) alloc(2048UL * 2);
  u16t*  eb3    = (u16t*) alloc(2048UL * 2);
  float* emo_f  = (float*)alloc(2048UL * 4);
  u16t*  mel_b  = (u16t*) alloc(327680UL * 2);
  u16t*  v1     = (u16t*) alloc(2097152UL * 2);
  u16t*  v2     = (u16t*) alloc(2097152UL * 2);
  u16t*  hbuf   = (u16t*)scores;                  // FFN hidden aliases scores
  if (used > ws_size) return;                     // workspace too small

  float* mel_out   = (float*)d_out;               // [8,512,80]
  float* wave_out  = mel_out + 327680;            // [8,512,256]
  float* logit_out = wave_out + 1048576;          // [8,8]

  // ============================ ENCODER ===================================
  embed_kernel<<<cdiv(1048576, 256), 256, 0, stream>>>(text, F(4), xe_f, xe_b, 1048576);

  for (int L = 0; L < 6; ++L) {
    int bse = 5 + L * 12;
    const float *qkv_w = F(bse+0), *qkv_b = F(bse+1), *o_w = F(bse+2), *o_b = F(bse+3);
    const float *f1w = F(bse+4), *f1b = F(bse+5), *f2w = F(bse+6), *f2b = F(bse+7);
    const float *ln1g = F(bse+8), *ln1b = F(bse+9), *ln2g = F(bse+10), *ln2b = F(bse+11);

    cvt(stream, qkv_w, wconv, nullptr, 1536L * 512);
    gemm(stream, xe_b, 512, 0,0, wconv, 512, 0,0,
         nullptr, qkvb, 1536, 0,0, 2048, 1536, 512, qkv_b, 0, 1, 1);
    vtrans_kernel<<<cdiv(1048576,256),256,0,stream>>>(qkvb, 1536, 1024, 64, 256, 8, vt, 1048576);
    // scores: z=(b,h), M=N=256, K=hd=64
    gemm(stream, qkvb, 1536, 256L*1536, 64, qkvb + 512, 1536, 256L*1536, 64,
         scores, nullptr, 256, 8L*65536, 65536, 256, 256, 64, nullptr, 0, 64, 8);
    softmax_kernel<<<64*256, 256, 0, stream>>>(scores, pbuf, 256, 0.125f);
    // P*V^T: M=256, N=hd=64, K=256
    gemm(stream, pbuf, 256, 8L*65536, 65536, vt, 256, 8L*16384, 16384,
         nullptr, attnb, 512, 256L*512, 64, 256, 64, 256, nullptr, 0, 64, 8);
    cvt(stream, o_w, wconv, nullptr, 512L * 512);
    gemm(stream, attnb, 512, 0,0, wconv, 512, 0,0,
         tmpf, nullptr, 512, 0,0, 2048, 512, 512, o_b, 0, 1, 1);
    add_ln_kernel<<<2048, 256, 0, stream>>>(xe_f, tmpf, ln1g, ln1b, xe_f, xe_b, 512);
    cvt(stream, f1w, wconv, nullptr, 2048L * 512);
    gemm(stream, xe_b, 512, 0,0, wconv, 512, 0,0,
         nullptr, hbuf, 2048, 0,0, 2048, 2048, 512, f1b, 1, 1, 1);
    cvt(stream, f2w, wconv, nullptr, 512L * 2048);
    gemm(stream, hbuf, 2048, 0,0, wconv, 2048, 0,0,
         tmpf, nullptr, 512, 0,0, 2048, 512, 2048, f2b, 0, 1, 1);
    add_ln_kernel<<<2048, 256, 0, stream>>>(xe_f, tmpf, ln2g, ln2b, xe_f, xe_b, 512);
  }

  // ====================== SPEAKER / EMOTION MLPs ==========================
  cvt(stream, F(1), sb_in, nullptr, 8L * 512);
  cvt(stream, F(77), wconv, nullptr, 512L * 512);
  gemm(stream, sb_in, 512, 0,0, wconv, 512, 0,0, nullptr, sb1, 512, 0,0, 8, 512, 512, F(78), 1, 1, 1);
  cvt(stream, F(79), wconv, nullptr, 512L * 512);
  gemm(stream, sb1, 512, 0,0, wconv, 512, 0,0, nullptr, sb2, 512, 0,0, 8, 512, 512, F(80), 1, 1, 1);
  cvt(stream, F(81), wconv, nullptr, 256L * 512);
  gemm(stream, sb2, 512, 0,0, wconv, 512, 0,0, spk_f, nullptr, 256, 0,0, 8, 256, 512, F(82), 0, 1, 1);

  cvt(stream, F(2), eb_in, nullptr, 8L * 256);
  cvt(stream, F(83), wconv, nullptr, 256L * 256);
  gemm(stream, eb_in, 256, 0,0, wconv, 256, 0,0, nullptr, eb1, 256, 0,0, 8, 256, 256, F(84), 1, 1, 1);
  cvt(stream, F(85), wconv, nullptr, 256L * 256);
  gemm(stream, eb1, 256, 0,0, wconv, 256, 0,0, nullptr, eb2, 256, 0,0, 8, 256, 256, F(86), 1, 1, 1);
  cvt(stream, F(87), wconv, nullptr, 256L * 256);
  gemm(stream, eb2, 256, 0,0, wconv, 256, 0,0, emo_f, eb3, 256, 0,0, 8, 256, 256, F(88), 0, 1, 1);
  // emo classifier -> logits straight into d_out
  cvt(stream, F(89), wconv, nullptr, 8L * 256);
  gemm(stream, eb3, 256, 0,0, wconv, 256, 0,0, logit_out, nullptr, 8, 0,0, 8, 8, 256, F(90), 0, 1, 1);

  // encoder memory: comb = [tf | spk | emo]  -> bf16 [2048,1024]
  comb_kernel<<<cdiv(2097152,256), 256, 0, stream>>>(xe_f, spk_f, emo_f, comb, 2097152);

  // ============================ DECODER ===================================
  cvt(stream, F(3), xd_b, xd_f, 4194304);   // x = target_mel (copy, don't mutate input)

  for (int L = 0; L < 6; ++L) {
    int bsd = 91 + L * 18;
    const float *sa_qkv_w = F(bsd+0),  *sa_qkv_b = F(bsd+1);
    const float *sa_o_w   = F(bsd+2),  *sa_o_b   = F(bsd+3);
    const float *ca_qkv_w = F(bsd+4),  *ca_qkv_b = F(bsd+5);
    const float *ca_o_w   = F(bsd+6),  *ca_o_b   = F(bsd+7);
    const float *f1w = F(bsd+8),  *f1b = F(bsd+9);
    const float *f2w = F(bsd+10), *f2b = F(bsd+11);
    const float *ln1g = F(bsd+12), *ln1b = F(bsd+13);
    const float *ln2g = F(bsd+14), *ln2b = F(bsd+15);
    const float *ln3g = F(bsd+16), *ln3b = F(bsd+17);

    // ---- self attention (S=512, D=1024, hd=128) ----
    cvt(stream, sa_qkv_w, wconv, nullptr, 3072L * 1024);
    gemm(stream, xd_b, 1024, 0,0, wconv, 1024, 0,0,
         nullptr, qkvb, 3072, 0,0, 4096, 3072, 1024, sa_qkv_b, 0, 1, 1);
    vtrans_kernel<<<cdiv(4194304,256),256,0,stream>>>(qkvb, 3072, 2048, 128, 512, 8, vt, 4194304);
    gemm(stream, qkvb, 3072, 512L*3072, 128, qkvb + 1024, 3072, 512L*3072, 128,
         scores, nullptr, 512, 8L*262144, 262144, 512, 512, 128, nullptr, 0, 64, 8);
    softmax_kernel<<<64*512, 256, 0, stream>>>(scores, pbuf, 512, 0.0883883476f);
    gemm(stream, pbuf, 512, 8L*262144, 262144, vt, 512, 8L*65536, 65536,
         nullptr, attnb, 1024, 512L*1024, 128, 512, 128, 512, nullptr, 0, 64, 8);
    cvt(stream, sa_o_w, wconv, nullptr, 1024L * 1024);
    gemm(stream, attnb, 1024, 0,0, wconv, 1024, 0,0,
         tmpf, nullptr, 1024, 0,0, 4096, 1024, 1024, sa_o_b, 0, 1, 1);
    add_ln_kernel<<<4096, 256, 0, stream>>>(xd_f, tmpf, ln1g, ln1b, xd_f, xd_b, 1024);

    // ---- cross attention (Sq=512, Sk=256) ----
    cvt(stream, ca_qkv_w, wconv, nullptr, 3072L * 1024);
    u16t* qb  = qkvb;
    u16t* kvb = qkvb + 4194304;
    gemm(stream, xd_b, 1024, 0,0, wconv, 1024, 0,0,
         nullptr, qb, 1024, 0,0, 4096, 1024, 1024, ca_qkv_b, 0, 1, 1);
    gemm(stream, comb, 1024, 0,0, wconv + 1024L*1024, 1024, 0,0,
         nullptr, kvb, 2048, 0,0, 2048, 2048, 1024, ca_qkv_b + 1024, 0, 1, 1);
    vtrans_kernel<<<cdiv(2097152,256),256,0,stream>>>(kvb, 2048, 1024, 128, 256, 8, vt, 2097152);
    gemm(stream, qb, 1024, 512L*1024, 128, kvb, 2048, 256L*2048, 128,
         scores, nullptr, 256, 8L*131072, 131072, 512, 256, 128, nullptr, 0, 64, 8);
    softmax_kernel<<<64*512, 256, 0, stream>>>(scores, pbuf, 256, 0.0883883476f);
    gemm(stream, pbuf, 256, 8L*131072, 131072, vt, 256, 8L*32768, 32768,
         nullptr, attnb, 1024, 512L*1024, 128, 512, 128, 256, nullptr, 0, 64, 8);
    cvt(stream, ca_o_w, wconv, nullptr, 1024L * 1024);
    gemm(stream, attnb, 1024, 0,0, wconv, 1024, 0,0,
         tmpf, nullptr, 1024, 0,0, 4096, 1024, 1024, ca_o_b, 0, 1, 1);
    add_ln_kernel<<<4096, 256, 0, stream>>>(xd_f, tmpf, ln2g, ln2b, xd_f, xd_b, 1024);

    // ---- FFN 1024 -> 4096 -> 1024 ----
    cvt(stream, f1w, wconv, nullptr, 4096L * 1024);
    gemm(stream, xd_b, 1024, 0,0, wconv, 1024, 0,0,
         nullptr, hbuf, 4096, 0,0, 4096, 4096, 1024, f1b, 1, 1, 1);
    cvt(stream, f2w, wconv, nullptr, 1024L * 4096);
    gemm(stream, hbuf, 4096, 0,0, wconv, 4096, 0,0,
         tmpf, nullptr, 1024, 0,0, 4096, 1024, 4096, f2b, 0, 1, 1);
    add_ln_kernel<<<4096, 256, 0, stream>>>(xd_f, tmpf, ln3g, ln3b, xd_f, xd_b, 1024);
  }

  // ====================== MEL HEAD + VOCODER ==============================
  cvt(stream, F(199), wconv, nullptr, 80L * 1024);
  gemm(stream, xd_b, 1024, 0,0, wconv, 1024, 0,0,
       mel_out, mel_b, 80, 0,0, 4096, 80, 1024, F(200), 0, 1, 1);
  cvt(stream, F(201), wconv, nullptr, 512L * 80);
  gemm(stream, mel_b, 80, 0,0, wconv, 80, 0,0,
       nullptr, v1, 512, 0,0, 4096, 512, 80, F(202), 1, 1, 1);
  cvt(stream, F(203), wconv, nullptr, 512L * 512);
  gemm(stream, v1, 512, 0,0, wconv, 512, 0,0,
       nullptr, v2, 512, 0,0, 4096, 512, 512, F(204), 1, 1, 1);
  cvt(stream, F(205), wconv, nullptr, 256L * 512);
  gemm(stream, v2, 512, 0,0, wconv, 512, 0,0,
       wave_out, nullptr, 256, 0,0, 4096, 256, 512, F(206), 0, 1, 1);
}
